// MarginLossLayer_66082366816434
// MI455X (gfx1250) — matrix operand order repaired and checked by reading, not compiled, not run on hardware
//
#include <hip/hip_runtime.h>
#include <stdint.h>

// ---------------------------------------------------------------------------
// Fused MarginLoss (distance-weighted sampling) for MI455X / gfx1250.
//
// Roofline: Gram = X @ X^T is 2*4096^2*512 = 17.2 GFLOP over 8 MB of input.
// X fits in L2 (192 MB), so HBM traffic is negligible -> matrix-pipe bound.
// We split f32 -> bf16 hi+lo and use v_wmma_f32_16x16x32_bf16 (4 WMMA per
// 32-k chunk: hihi+hilo+lohi+lolo ~= f32 accuracy at bf16 WMMA rate) instead
// of the 16x16x4 f32 WMMA (which would need 8x more issues).
// The 67 MB distance matrix is never materialized: each 16x16 tile is
// consumed immediately (hinge sums + threefry2x32 Gumbel-max categorical
// sampling bit-matching jax.random.categorical(key=42)).
// Occupancy is launch-limited (~2 waves/SIMD), so the next j-tile's B rows
// are prefetched (global_prefetch_b8) a full k-loop (64 WMMAs) ahead.
// ---------------------------------------------------------------------------

#define B_N       4096
#define D_K       512
#define LDA       (D_K + 8)       // LDS row padding (shorts) to spread banks
#define NTILE     (B_N / 16)      // 256
#define ALPHA_C   0.2f
#define CUTOFF_C  0.5f
#define NU_C      0.0f
#define TINY_F    1.17549435e-38f

typedef __attribute__((ext_vector_type(16))) __bf16 v16bf;
typedef __attribute__((ext_vector_type(8)))  float  v8f;

// workspace layout (bytes)
#define WS_XHI   0
#define WS_XLO   (B_N * D_K * 2)
#define WS_G     (WS_XLO + B_N * D_K * 2)
#define WS_PART  (WS_G + B_N * 4)
#define WS_TOTAL (WS_PART + NTILE * 8 * 4)

__device__ __forceinline__ unsigned short f32_to_bf16_rne(float f) {
  unsigned u = __float_as_uint(f);
  u += 0x7fffu + ((u >> 16) & 1u);
  return (unsigned short)(u >> 16);
}
__device__ __forceinline__ float bf16_to_f32(unsigned short h) {
  return __uint_as_float(((unsigned)h) << 16);
}

// Exact replica of JAX threefry2x32 random bits for key(42), flat array of
// size 7*4096*4096 (even): element idx < half -> out0 of counters (idx,
// idx+half), else out1 of (idx-half, idx).
__device__ __forceinline__ unsigned tf_bits(unsigned idx) {
  const unsigned HALF = 58720256u;  // 7*4096*4096/2
  unsigned x0, x1, sel;
  if (idx < HALF) { x0 = idx;        x1 = idx + HALF; sel = 0u; }
  else            { x0 = idx - HALF; x1 = idx;        sel = 1u; }
  const unsigned k0 = 0u, k1 = 42u, k2 = 0x1BD11BDAu ^ 0u ^ 42u;
  x0 += k0; x1 += k1;
#define TF_ROUND(r) { x0 += x1; x1 = (x1 << (r)) | (x1 >> (32 - (r))); x1 ^= x0; }
  TF_ROUND(13) TF_ROUND(15) TF_ROUND(26) TF_ROUND(6)
  x0 += k1; x1 += k2 + 1u;
  TF_ROUND(17) TF_ROUND(29) TF_ROUND(16) TF_ROUND(24)
  x0 += k2; x1 += k0 + 2u;
  TF_ROUND(13) TF_ROUND(15) TF_ROUND(26) TF_ROUND(6)
  x0 += k0; x1 += k1 + 3u;
  TF_ROUND(17) TF_ROUND(29) TF_ROUND(16) TF_ROUND(24)
  x0 += k1; x1 += k2 + 4u;
  TF_ROUND(13) TF_ROUND(15) TF_ROUND(26) TF_ROUND(6)
  x0 += k2; x1 += k0 + 5u;
#undef TF_ROUND
  return sel ? x1 : x0;
}

// A-matrix 16-bit fragment (16x32): lane holds row M=lane&15; elements 0..7
// are K kA..kA+7, elements 8..15 are K kA+16..kA+23 (kA already includes the
// lane-half offset 0/8). Two ds_load_b128.
__device__ __forceinline__ v16bf frag_a(const unsigned short* p) {
  union { uint4 q[2]; v16bf v; } u;
  u.q[0] = *(const uint4*)(p);
  u.q[1] = *(const uint4*)(p + 16);
  return u.v;
}
// B-matrix 16-bit fragment (32x16): lane holds col N=lane&15; elements 0..7
// are K kB..kB+7, elements 8..15 are K kB+16..kB+23 (kB includes lane-half
// offset 0/8). Two b128 loads.
__device__ __forceinline__ v16bf frag_b(const unsigned short* p) {
  union { uint4 q[2]; v16bf v; } u;
  u.q[0] = *(const uint4*)(p);
  u.q[1] = *(const uint4*)(p + 16);
  return u.v;
}

// ---------------------------------------------------------------------------
// Prep: Dekker-split f32 embedding into bf16 hi/lo, compute g[i] = |x_i|^2.
// ---------------------------------------------------------------------------
__global__ __launch_bounds__(256)
void marginloss_prep(const float* __restrict__ emb,
                     unsigned short* __restrict__ xhi,
                     unsigned short* __restrict__ xlo,
                     float* __restrict__ g) {
  __shared__ float red[256];
  const int row = blockIdx.x;
  const int t = threadIdx.x;
  float s = 0.f;
  for (int k = t; k < D_K; k += 256) {
    const float x = emb[(size_t)row * D_K + k];
    s += x * x;
    const unsigned short h = f32_to_bf16_rne(x);
    const unsigned short l = f32_to_bf16_rne(x - bf16_to_f32(h));
    xhi[(size_t)row * D_K + k] = h;
    xlo[(size_t)row * D_K + k] = l;
  }
  red[t] = s;
  __syncthreads();
  for (int off = 128; off > 0; off >>= 1) {
    if (t < off) red[t] += red[t + off];
    __syncthreads();
  }
  if (t == 0) g[row] = red[0];
}

// ---------------------------------------------------------------------------
// Main fused kernel: 1 WG = 16 anchor rows, 4 waves split the 256 j-tiles.
// ---------------------------------------------------------------------------
__global__ __launch_bounds__(128)
void marginloss_main(const unsigned short* __restrict__ xhi,
                     const unsigned short* __restrict__ xlo,
                     const float* __restrict__ g,
                     const int* __restrict__ labels,
                     const float* __restrict__ betas,
                     float* __restrict__ part) {
  __shared__ unsigned short sAhi[16 * LDA];
  __shared__ unsigned short sAlo[16 * LDA];
  __shared__ float sGi[16];
  __shared__ float sD[4][16][16];
  __shared__ int   sLab[4][16];
  __shared__ float rVal[4][7][16];
  __shared__ float rDst[4][7][16];
  __shared__ float pPart[4][16][5];
  __shared__ float fin[16][6];

  const int tid = threadIdx.x;
  const int i0 = blockIdx.x * 16;

  // Stage anchor tile (hi+lo) into LDS (b128 granularity).
  for (int t = tid; t < 16 * (D_K / 8); t += 128) {
    const int r = t >> 6;            // 64 8-short chunks per row
    const int c = (t & 63) << 3;
    *(uint4*)&sAhi[r * LDA + c] = *(const uint4*)&xhi[(size_t)(i0 + r) * D_K + c];
    *(uint4*)&sAlo[r * LDA + c] = *(const uint4*)&xlo[(size_t)(i0 + r) * D_K + c];
  }
  if (tid < 16) sGi[tid] = g[i0 + tid];
  __syncthreads();

  const int wave = tid >> 5;
  const int lane = tid & 31;
  const int rn = lane & 15;   // A-row index == B-col index == sampling row
  const int sg = lane >> 4;   // lane half (selects K groups and s-subset)

  const int   iGlob  = i0 + rn;
  const int   li     = labels[iGlob];
  const float beta_i = betas[li];
  const float margin = beta_i + ALPHA_C;

  const int nS    = (sg == 0) ? 4 : 3;  // s in {0..3} / {4..6}
  const int sBase = (sg == 0) ? 0 : 4;

  float bestV[4], bestD[4];
#pragma unroll
  for (int s = 0; s < 4; ++s) { bestV[s] = -__builtin_inff(); bestD[s] = 0.f; }
  float posl = 0.f, pairs = 0.f, npos = 0.f, cnt = 0.f, hasfin = 0.f;

  const int kAoff = sg ? 8 : 0;
  const int kBoff = sg ? 8 : 0;
  const unsigned short* aHiRow = &sAhi[rn * LDA];
  const unsigned short* aLoRow = &sAlo[rn * LDA];

  for (int jt = wave; jt < NTILE; jt += 4) {
    const int j0 = jt * 16;
    const unsigned short* bHiRow = xhi + (size_t)(j0 + rn) * D_K;
    const unsigned short* bLoRow = xlo + (size_t)(j0 + rn) * D_K;

    // Prefetch this wave's NEXT j-tile rows a full k-loop (64 WMMAs) ahead:
    // lowers to global_prefetch_b8, no LOADcnt cost, hides the L2 round-trip
    // at our launch-limited (~2 waves/SIMD) occupancy.
    if (jt + 4 < NTILE) {
      const unsigned short* nHi = xhi + (size_t)(j0 + 64 + rn) * D_K;
      const unsigned short* nLo = xlo + (size_t)(j0 + 64 + rn) * D_K;
#pragma unroll
      for (int pf = 0; pf < 4; ++pf) {
        __builtin_prefetch(nHi + pf * 128, 0, 3);   // 4 x 256B per 1KB row
        __builtin_prefetch(nLo + pf * 128, 0, 3);
      }
    }

    v8f acc0 = {};
    v8f acc1 = {};
#pragma unroll 4
    for (int c = 0; c < 16; ++c) {
      const int kA = c * 32 + kAoff;
      const int kB = c * 32 + kBoff;
      const v16bf ah = frag_a(aHiRow + kA);
      const v16bf al = frag_a(aLoRow + kA);
      const v16bf bh = frag_b(bHiRow + kB);
      const v16bf bl = frag_b(bLoRow + kB);
      // two independent accumulator chains for ILP
      acc0 = __builtin_amdgcn_wmma_f32_16x16x32_bf16(false, al, false, bl, (short)0, acc0, false, false);
      acc1 = __builtin_amdgcn_wmma_f32_16x16x32_bf16(false, al, false, bh, (short)0, acc1, false, false);
      acc1 = __builtin_amdgcn_wmma_f32_16x16x32_bf16(false, ah, false, bl, (short)0, acc1, false, false);
      acc0 = __builtin_amdgcn_wmma_f32_16x16x32_bf16(false, ah, false, bh, (short)0, acc0, false, false);
    }

    // d tile: lane holds (m = e + 8*sg, n = rn) of C/D per the ISA layout.
    const float gj = g[j0 + rn];
#pragma unroll
    for (int e = 0; e < 8; ++e) {
      const int m = e + 8 * sg;
      const float dot = acc0[e] + acc1[e];
      const float d2 = sGi[m] + gj - 2.0f * dot;
      sD[wave][m][rn] = sqrtf(fmaxf(d2, 0.05f));
    }
    if (sg == 0) sLab[wave][rn] = labels[j0 + rn];
    asm volatile("s_wait_dscnt 0" ::: "memory");  // in-wave LDS RAW visibility

    // Row-per-lane pass: lane owns anchor row rn; sg splits the 7 samples.
    for (int t = 0; t < 16; ++t) {
      const int   j  = j0 + t;
      const float d  = sD[wave][rn][t];
      const int   lj = sLab[wave][t];
      const bool same = (lj == li);
      if (sg == 0 && same) {
        cnt += 1.f;                         // count[i] includes self
        if (j != iGlob) {
          const float pl = fmaxf(ALPHA_C + d - beta_i, 0.f);
          posl  += pl;
          pairs += (pl > 0.f) ? 1.f : 0.f;
          npos  += 1.f;                     // beta_ap counts all pm pairs
        }
      }
      if (!same && d < margin) {
        const float dd = fmaxf(d, CUTOFF_C);
        // (2-n)log d - (n-3)/2 log(1-d^2/4), n=512. Row-max shift skipped:
        // argmax is invariant to per-row constants.
        const float lq = -510.0f * logf(dd) - 254.5f * logf(1.0f - 0.25f * dd * dd);
        if (sg == 0) hasfin = 1.f;
        for (int ss = 0; ss < nS; ++ss) {
          const unsigned idx = ((unsigned)(sBase + ss) << 24) |
                               ((unsigned)iGlob << 12) | (unsigned)j;
          const unsigned bits = tf_bits(idx);
          float u = __uint_as_float((bits >> 9) | 0x3f800000u) - 1.0f;
          u = fmaxf(TINY_F, u * (1.0f - TINY_F) + TINY_F);
          const float score = lq - logf(-logf(u));   // logit + gumbel
          if (score > bestV[ss]) { bestV[ss] = score; bestD[ss] = d; }
        }
      }
    }
  }

  // Per-wave partials -> LDS.
  for (int ss = 0; ss < nS; ++ss) {
    rVal[wave][sBase + ss][rn] = bestV[ss];
    rDst[wave][sBase + ss][rn] = bestD[ss];
  }
  if (sg == 0) {
    pPart[wave][rn][0] = posl;
    pPart[wave][rn][1] = pairs;
    pPart[wave][rn][2] = npos;
    pPart[wave][rn][3] = cnt;
    pPart[wave][rn][4] = hasfin;
  }
  __syncthreads();

  // Combine 4 waves, finish per-row negative terms.
  if (tid < 16) {
    const int row = tid;
    const float bi = betas[labels[i0 + row]];
    float Pl = 0.f, Pp = 0.f, Np = 0.f, Ct = 0.f, Hf = 0.f;
    for (int w = 0; w < 4; ++w) {
      Pl += pPart[w][row][0]; Pp += pPart[w][row][1];
      Np += pPart[w][row][2]; Ct += pPart[w][row][3];
      Hf += pPart[w][row][4];
    }
    float negl = 0.f, negp = 0.f, ban = 0.f;
    const float toS = Ct - 1.f;
    for (int s = 0; s < 7; ++s) {
      float bv = -__builtin_inff(), bd = 0.f;
      for (int w = 0; w < 4; ++w) {
        const float v = rVal[w][s][row];
        if (v > bv) { bv = v; bd = rDst[w][s][row]; }
      }
      if (((float)s < toS) && (Hf > 0.f) && (bv > -__builtin_inff())) {
        const float nl = fmaxf(ALPHA_C + bi - bd, 0.f);
        negl += nl;
        negp += (nl > 0.f) ? 1.f : 0.f;
        ban  += bi;
      }
    }
    fin[row][0] = Pl;       fin[row][1] = Pp;
    fin[row][2] = negl;     fin[row][3] = negp;
    fin[row][4] = bi * Np;  fin[row][5] = ban;
  }
  __syncthreads();
  if (tid == 0) {
    float a0 = 0.f, a1 = 0.f, a2 = 0.f, a3 = 0.f, a4 = 0.f, a5 = 0.f;
    for (int r = 0; r < 16; ++r) {
      a0 += fin[r][0]; a1 += fin[r][1]; a2 += fin[r][2];
      a3 += fin[r][3]; a4 += fin[r][4]; a5 += fin[r][5];
    }
    float* p = part + (size_t)blockIdx.x * 8;
    p[0] = a0; p[1] = a1; p[2] = a2; p[3] = a3; p[4] = a4; p[5] = a5;
    p[6] = 0.f; p[7] = 0.f;
  }
}

// ---------------------------------------------------------------------------
// Finalize: deterministic fixed-order sum over the 256 WG partials.
// ---------------------------------------------------------------------------
__global__ void marginloss_finalize(const float* __restrict__ part,
                                    float* __restrict__ out) {
  if (blockIdx.x == 0 && threadIdx.x == 0) {
    float pl = 0.f, pp = 0.f, nl = 0.f, np = 0.f, bap = 0.f, ban = 0.f;
    for (int b = 0; b < NTILE; ++b) {
      const float* p = part + (size_t)b * 8;
      pl += p[0]; pp += p[1]; nl += p[2]; np += p[3]; bap += p[4]; ban += p[5];
    }
    const float beta_loss = fmaxf((bap + ban) * NU_C, 0.f);
    const float prs = fmaxf(pp + np, 1.f);
    out[0] = (pl + nl + beta_loss) / prs;
  }
}

extern "C" void kernel_launch(void* const* d_in, const int* in_sizes, int n_in,
                              void* d_out, int out_size, void* d_ws, size_t ws_size,
                              hipStream_t stream) {
  (void)in_sizes; (void)n_in; (void)out_size;
  if (ws_size < (size_t)WS_TOTAL) return;  // need ~8.4 MB scratch

  const float* emb    = (const float*)d_in[0];
  const float* betas  = (const float*)d_in[1];
  const int*   labels = (const int*)d_in[2];

  char* ws = (char*)d_ws;
  unsigned short* xhi = (unsigned short*)(ws + WS_XHI);
  unsigned short* xlo = (unsigned short*)(ws + WS_XLO);
  float* g    = (float*)(ws + WS_G);
  float* part = (float*)(ws + WS_PART);
  float* out  = (float*)d_out;

  hipLaunchKernelGGL(marginloss_prep, dim3(B_N), dim3(256), 0, stream,
                     emb, xhi, xlo, g);
  hipLaunchKernelGGL(marginloss_main, dim3(NTILE), dim3(128), 0, stream,
                     xhi, xlo, g, labels, betas, part);
  hipLaunchKernelGGL(marginloss_finalize, dim3(1), dim3(1), 0, stream,
                     part, out);
}
